// SecondOrderDeformableAlignment_67980742361829
// MI455X (gfx1250) — compile-verified, matrix operand order checked by reading, not run on hardware
//
#include <hip/hip_runtime.h>
#include <hip/hip_bf16.h>

// ---------------------------------------------------------------------------
// CDNA5 / gfx1250 implementation of second-order deformable alignment.
// All GEMM-shaped work lowered to v_wmma_f32_16x16x32_bf16 (f32 accumulate),
// two independent accumulators per wave for XDL pipelining, compile-time
// tile geometry (no runtime div in inner loops), async global->LDS staging.
// ---------------------------------------------------------------------------

typedef __attribute__((ext_vector_type(16))) __bf16 v16bf;
typedef __attribute__((ext_vector_type(8)))  __bf16 v8bf;
typedef __attribute__((ext_vector_type(8)))  float  v8f;

// Exact parameter types of __builtin_amdgcn_global_load_async_to_lds_b128:
//   (v4i addrspace(1)*, v4i addrspace(3)*, imm int, imm int)
typedef int v4i_g __attribute__((vector_size(16)));
typedef __attribute__((address_space(1))) v4i_g* as1_v4i_ptr;
typedef __attribute__((address_space(3))) v4i_g* as3_v4i_ptr;

union V16 { v16bf v; v8bf h[2]; };
union U8B { v8bf v; unsigned int u[4]; };

#define HD __device__ __forceinline__

constexpr int B_   = 4;
constexpr int H_   = 96;
constexpr int W_   = 96;
constexpr int HW   = H_ * W_;
constexpr int CIN  = 128;
constexpr int COUT = 64;
constexpr int COFF = 196;   // 3*Cout + 4
constexpr int COFFP= 224;   // padded to multiple of 32
constexpr int C3O  = 432;   // 27*DG
constexpr int KD   = 1152;  // deform K = G*K2*Cg = 16*9*8

#if defined(__gfx1250__) && __has_builtin(__builtin_amdgcn_global_load_async_to_lds_b128) && __has_builtin(__builtin_amdgcn_s_wait_asynccnt)
#define HAVE_ASYNC_LDS 1
#endif

HD v8f wmma_bf16(v16bf a, v16bf b, v8f c) {
#if defined(__gfx1250__)
  return __builtin_amdgcn_wmma_f32_16x16x32_bf16(false, a, false, b, (short)0, c,
                                                 false, false);
#else
  return c;
#endif
}

// ---------------------------------------------------------------------------
// Build padded NHWC bf16 feature tensor: feat[b][y][x][c], c in [0,224)
// c<192: extra_feat ; 192/193: flow_1 ; 194/195: flow_2 ; else 0
// ---------------------------------------------------------------------------
__global__ void k_pack_feat(const float* __restrict__ extra,
                            const float* __restrict__ f1,
                            const float* __restrict__ f2,
                            __bf16* __restrict__ feat) {
  const int total = B_ * HW * COFFP;
  for (int idx = blockIdx.x * blockDim.x + threadIdx.x; idx < total;
       idx += gridDim.x * blockDim.x) {
    int c  = idx % COFFP;
    int sp = idx / COFFP;
    int b  = sp / HW;
    int yx = sp % HW;
    float v = 0.f;
    if (c < 192)      v = extra[(b * 192 + c) * HW + yx];
    else if (c < 194) v = f1[(b * 2 + (c - 192)) * HW + yx];
    else if (c < 196) v = f2[(b * 2 + (c - 194)) * HW + yx];
    feat[idx] = (__bf16)v;
  }
}

// ---------------------------------------------------------------------------
// Repack OIHW f32 weights into WMMA B-fragment order (bf16).
// K-chunk q (32 wide), lane l, slot i: per ISA 16-bit operand layout,
//   kk = (i<8 ? i : i+8) + (l>=16 ? 8 : 0),  n = ntile*16 + (l&15)
// Conv mapping  (deform=0): k = tap*Cpad + c            -> w[(n*C+c)*9 + tap]
// Deform mapping(deform=1): k = (g*9+t)*8 + c, cin=g*8+c -> w[(n*128+cin)*9+t]
// ---------------------------------------------------------------------------
__global__ void k_repack(const float* __restrict__ w, __bf16* __restrict__ pw,
                         int Nreal, int Creal, int Cpad, int ntiles,
                         int nchunks, int deform) {
  const int total = ntiles * nchunks * 512;
  for (int idx = blockIdx.x * blockDim.x + threadIdx.x; idx < total;
       idx += gridDim.x * blockDim.x) {
    int i  = idx & 15;
    int l  = (idx >> 4) & 31;
    int q  = (idx >> 9) % nchunks;
    int nt = idx / (512 * nchunks);
    int kk = (i < 8 ? i : i + 8) + ((l >= 16) ? 8 : 0);
    int k  = q * 32 + kk;
    int n  = nt * 16 + (l & 15);
    float val = 0.f;
    if (!deform) {
      int tap = k / Cpad, c = k % Cpad;
      if (n < Nreal && c < Creal) val = w[(n * Creal + c) * 9 + tap];
    } else {
      int g = k / 72, r = k % 72, t = r >> 3, c = r & 7;
      int cin = g * 8 + c;
      if (n < Nreal) val = w[(n * CIN + cin) * 9 + t];
    }
    pw[idx] = (__bf16)val;
  }
}

// ---------------------------------------------------------------------------
// Stage (3 rows x 34 cols x CPAD ch) NHWC halo patch into LDS, layout
// lds[(r*36 + col)*CPAD + c]  (contiguous in c -> b128 on both sides).
// ---------------------------------------------------------------------------
template <int CPAD>
HD void load_patch(const __bf16* __restrict__ in, __bf16* lds, int b, int y,
                   int x0, int tid, int nthreads) {
  constexpr int C8    = CPAD >> 3;
  constexpr int units = 3 * 34 * C8;
  for (int u = tid; u < units; u += nthreads) {
    int c8 = u % C8;
    int cc = (u / C8) % 34;
    int rr = u / (C8 * 34);
    int yy = y + rr - 1;
    int xx = x0 - 1 + cc;
    int dst = (rr * 36 + cc) * CPAD + c8 * 8;
    bool valid = (yy >= 0) && (yy < H_) && (xx >= 0) && (xx < W_);
    if (valid) {
      int src = (((b * H_ + yy) * W_) + xx) * CPAD + c8 * 8;
#if defined(HAVE_ASYNC_LDS)
      __builtin_amdgcn_global_load_async_to_lds_b128(
          (as1_v4i_ptr)(in + src), (as3_v4i_ptr)(lds + dst), 0, 0);
#else
      *(v8bf*)&lds[dst] = *(const v8bf*)&in[src];
#endif
    } else {
      U8B z; z.u[0] = z.u[1] = z.u[2] = z.u[3] = 0u;
      *(v8bf*)&lds[dst] = z.v;
    }
  }
#if defined(HAVE_ASYNC_LDS)
  __builtin_amdgcn_s_wait_asynccnt(0);
#endif
  __syncthreads();
}

// Build 16x32 bf16 A-fragment for chunk q from the LDS patch (CPAD%32==0,
// so with compile-time CPAD all div/mod fold to shifts/constants).
template <int CPAD>
HD v16bf a_frag(const __bf16* lds, int q, int msub, int l) {
  int tap   = (q * 32) / CPAD;
  int cbase = (q * 32) % CPAD;
  int r  = tap / 3, dx = tap % 3;
  int hi = (l >> 4) & 1;
  int col   = msub * 16 + (l & 15) + dx;
  int abase = (r * 36 + col) * CPAD + cbase + hi * 8;
  V16 a;
  a.h[0] = *(const v8bf*)&lds[abase];
  a.h[1] = *(const v8bf*)&lds[abase + 16];
  return a.v;
}

// ---------------------------------------------------------------------------
// 3x3 conv + bias + leaky-relu, implicit GEMM.  Block = (b, y, 32-wide x tile),
// 128 threads = 4 waves.  Wave w: msub = w&1 (x half), two n-tiles
// {2*(w>>1), 2*(w>>1)+1} -> two independent accumulators sharing one A-frag.
// Output NHWC bf16 with 64 channels.
// ---------------------------------------------------------------------------
template <int CPAD, int NCHUNKS>
__global__ void k_conv_lrelu(const __bf16* __restrict__ in,
                             const __bf16* __restrict__ pw,
                             const float* __restrict__ bias,
                             __bf16* __restrict__ out) {
  extern __shared__ __bf16 lds[];
  int blk = blockIdx.x;
  int xb = blk % 3, y = (blk / 3) % H_, b = blk / (3 * H_);
  int x0 = xb * 32;
  int tid = threadIdx.x;
  load_patch<CPAD>(in, lds, b, y, x0, tid, blockDim.x);

  int l = tid & 31, w = tid >> 5;
  int msub = w & 1;
  int nt0 = (w >> 1) * 2, nt1 = nt0 + 1;
  int hi = (l >> 4) & 1;
  v8f acc0 = {0.f, 0.f, 0.f, 0.f, 0.f, 0.f, 0.f, 0.f};
  v8f acc1 = acc0;
  for (int q = 0; q < NCHUNKS; ++q) {
    v16bf a = a_frag<CPAD>(lds, q, msub, l);
    V16 b0; b0.v = *(const v16bf*)&pw[((nt0 * NCHUNKS + q) * 32 + l) * 16];
    V16 b1; b1.v = *(const v16bf*)&pw[((nt1 * NCHUNKS + q) * 32 + l) * 16];
    acc0 = wmma_bf16(a, b0.v, acc0);
    acc1 = wmma_bf16(a, b1.v, acc1);
  }
  int xbase = x0 + msub * 16;
  int n0 = nt0 * 16 + (l & 15), n1 = n0 + 16;
  float bn0 = bias[n0], bn1 = bias[n1];
  for (int v = 0; v < 8; ++v) {
    int m = v + hi * 8;
    int sp = ((b * H_ + y) * W_) + xbase + m;
    float v0 = acc0[v] + bn0;
    float v1 = acc1[v] + bn1;
    v0 = (v0 >= 0.f) ? v0 : 0.1f * v0;
    v1 = (v1 >= 0.f) ? v1 : 0.1f * v1;
    out[sp * COUT + n0] = (__bf16)v0;
    out[sp * COUT + n1] = (__bf16)v1;
  }
}

// ---------------------------------------------------------------------------
// conv3 (N=432, padded to 28 n-tiles) fused with offset/mask epilogue:
//   c<288 : offs = 10*tanh(y) + flow_{1|2}[1-(c&1)]   -> offset[sp][288]
//   c>=288: mask = sigmoid(y)                         -> mask[sp][144]
// 128 threads, wave w: msub=w&1, n-tiles blockIdx.y*4 + 2*(w>>1) + {0,1}.
// ---------------------------------------------------------------------------
__global__ void k_conv_offsets(const __bf16* __restrict__ in,
                               const __bf16* __restrict__ pw,
                               const float* __restrict__ bias,
                               const float* __restrict__ f1,
                               const float* __restrict__ f2,
                               float* __restrict__ offs,
                               float* __restrict__ mask) {
  extern __shared__ __bf16 lds[];
  constexpr int CPAD = 64, NCHUNKS = 18;
  int blk = blockIdx.x;
  int xb = blk % 3, y = (blk / 3) % H_, b = blk / (3 * H_);
  int x0 = xb * 32;
  int tid = threadIdx.x;
  load_patch<CPAD>(in, lds, b, y, x0, tid, blockDim.x);

  int l = tid & 31, w = tid >> 5;
  int msub = w & 1;
  int nt0 = blockIdx.y * 4 + (w >> 1) * 2, nt1 = nt0 + 1;  // < 28 (padded)
  int hi = (l >> 4) & 1;
  v8f acc0 = {0.f, 0.f, 0.f, 0.f, 0.f, 0.f, 0.f, 0.f};
  v8f acc1 = acc0;
  for (int q = 0; q < NCHUNKS; ++q) {
    v16bf a = a_frag<CPAD>(lds, q, msub, l);
    V16 b0; b0.v = *(const v16bf*)&pw[((nt0 * NCHUNKS + q) * 32 + l) * 16];
    V16 b1; b1.v = *(const v16bf*)&pw[((nt1 * NCHUNKS + q) * 32 + l) * 16];
    acc0 = wmma_bf16(a, b0.v, acc0);
    acc1 = wmma_bf16(a, b1.v, acc1);
  }
  int xbase = x0 + msub * 16;
  for (int half = 0; half < 2; ++half) {
    int nt = half ? nt1 : nt0;
    if (nt >= 27) continue;  // wave-uniform guard (padding tile)
    const v8f& acc = half ? acc1 : acc0;
    int n = nt * 16 + (l & 15);
    float bn = bias[n];
    for (int v = 0; v < 8; ++v) {
      int m = v + hi * 8;
      int x = xbase + m;
      int sp = (b * H_ + y) * W_ + x;
      float val = acc[v] + bn;
      if (n < 288) {
        const float* fl = (n < 144) ? f1 : f2;
        int fch = 1 - (n & 1);  // flow[:, ::-1] tiled over channel pairs
        offs[sp * 288 + n] =
            10.f * tanhf(val) + fl[(b * 2 + fch) * HW + y * W_ + x];
      } else {
        mask[sp * 144 + (n - 288)] = 1.f / (1.f + expf(-val));
      }
    }
  }
}

// ---------------------------------------------------------------------------
// Deformable conv: per block (b, y, 32-wide x tile), 128 threads (4 waves).
// Phase 1: bilinear-sample + mask -> LDS sampA[m][k], k=(g*9+t)*8+c, bf16,
//          m in [0,32).
// Phase 2: wave w: msub=w&1, n-tiles {2*(w>>1), 2*(w>>1)+1}; 36 WMMA chunks
//          each against repacked deform weight; f32 NCHW output.
// ---------------------------------------------------------------------------
__global__ void k_deform(const float* __restrict__ x,
                         const float* __restrict__ offs,
                         const float* __restrict__ mask,
                         const __bf16* __restrict__ pwd,
                         const float* __restrict__ bias,
                         float* __restrict__ out) {
  extern __shared__ __bf16 sampA[];  // 32 * 1152 bf16 = 72 KB
  int blk = blockIdx.x;
  int xb = blk % 3, y = (blk / 3) % H_, b = blk / (3 * H_);
  int x0 = xb * 32;
  int tid = threadIdx.x;

  for (int u = tid; u < 32 * 144; u += blockDim.x) {
    int gt = u % 144, m = u / 144;
    int g = gt / 9, t = gt % 9;
    int xx = x0 + m;
    int sp = (b * H_ + y) * W_ + xx;
    float dy = offs[sp * 288 + gt * 2 + 0];
    float dx = offs[sp * 288 + gt * 2 + 1];
    float mk = mask[sp * 144 + gt];
    float py = dy + (float)(t / 3) + (float)y - 1.f;
    float px = dx + (float)(t % 3) + (float)xx - 1.f;
    float y0f = floorf(py), x0f = floorf(px);
    float wy1 = py - y0f, wx1 = px - x0f;
    int yi0 = (int)y0f, xi0 = (int)x0f;
    float wgt[4]; int idxv[4];
    int k = 0;
    for (int ty = 0; ty < 2; ++ty)
      for (int tx = 0; tx < 2; ++tx) {
        int yy = yi0 + ty, xv = xi0 + tx;
        float wv = (ty ? wy1 : 1.f - wy1) * (tx ? wx1 : 1.f - wx1);
        bool valid = (yy >= 0) && (yy <= H_ - 1) && (xv >= 0) && (xv <= W_ - 1);
        int yc = yy < 0 ? 0 : (yy > H_ - 1 ? H_ - 1 : yy);
        int xc = xv < 0 ? 0 : (xv > W_ - 1 ? W_ - 1 : xv);
        wgt[k]  = valid ? wv : 0.f;
        idxv[k] = yc * W_ + xc;
        ++k;
      }
    const float* xbase = x + (b * CIN + g * 8) * HW;
    for (int c = 0; c < 8; ++c) {
      const float* xc = xbase + c * HW;
      float s = wgt[0] * xc[idxv[0]] + wgt[1] * xc[idxv[1]] +
                wgt[2] * xc[idxv[2]] + wgt[3] * xc[idxv[3]];
      sampA[m * KD + gt * 8 + c] = (__bf16)(s * mk);
    }
  }
  __syncthreads();

  int l = tid & 31, w = tid >> 5;  // 4 waves
  int msub = w & 1;
  int nt0 = (w >> 1) * 2, nt1 = nt0 + 1;
  int hi = (l >> 4) & 1;
  v8f acc0 = {0.f, 0.f, 0.f, 0.f, 0.f, 0.f, 0.f, 0.f};
  v8f acc1 = acc0;
  for (int q = 0; q < 36; ++q) {
    int abase = (msub * 16 + (l & 15)) * KD + q * 32 + hi * 8;
    V16 a;
    a.h[0] = *(const v8bf*)&sampA[abase];
    a.h[1] = *(const v8bf*)&sampA[abase + 16];
    V16 b0; b0.v = *(const v16bf*)&pwd[((nt0 * 36 + q) * 32 + l) * 16];
    V16 b1; b1.v = *(const v16bf*)&pwd[((nt1 * 36 + q) * 32 + l) * 16];
    acc0 = wmma_bf16(a.v, b0.v, acc0);
    acc1 = wmma_bf16(a.v, b1.v, acc1);
  }
  int n0 = nt0 * 16 + (l & 15), n1 = n0 + 16;
  float bn0 = bias[n0], bn1 = bias[n1];
  int xbase = x0 + msub * 16;
  for (int v = 0; v < 8; ++v) {
    int m = v + hi * 8;
    out[((b * COUT + n0) * H_ + y) * W_ + xbase + m] = acc0[v] + bn0;
    out[((b * COUT + n1) * H_ + y) * W_ + xbase + m] = acc1[v] + bn1;
  }
}

// ---------------------------------------------------------------------------
// Host launcher
// ---------------------------------------------------------------------------
extern "C" void kernel_launch(void* const* d_in, const int* in_sizes, int n_in,
                              void* d_out, int out_size, void* d_ws,
                              size_t ws_size, hipStream_t stream) {
  const float* x     = (const float*)d_in[0];
  const float* extra = (const float*)d_in[1];
  const float* f1    = (const float*)d_in[2];
  const float* f2    = (const float*)d_in[3];
  const float* wgt   = (const float*)d_in[4];
  const float* bias  = (const float*)d_in[5];
  const float* ow0   = (const float*)d_in[6];
  const float* ob0   = (const float*)d_in[7];
  const float* ow1   = (const float*)d_in[8];
  const float* ob1   = (const float*)d_in[9];
  const float* ow2   = (const float*)d_in[10];
  const float* ob2   = (const float*)d_in[11];
  const float* ow3   = (const float*)d_in[12];
  const float* ob3   = (const float*)d_in[13];
  float* out = (float*)d_out;

  size_t off = 0;
  auto carve = [&](size_t bytes) -> void* {
    void* p = (char*)d_ws + off;
    off += (bytes + 255) & ~(size_t)255;
    return p;
  };
  __bf16* feat  = (__bf16*)carve((size_t)B_ * HW * COFFP * 2);
  __bf16* h1    = (__bf16*)carve((size_t)B_ * HW * COUT * 2);
  __bf16* h2    = (__bf16*)carve((size_t)B_ * HW * COUT * 2);
  __bf16* h3    = (__bf16*)carve((size_t)B_ * HW * COUT * 2);
  float*  offsb = (float*)carve((size_t)B_ * HW * 288 * 4);
  float*  maskb = (float*)carve((size_t)B_ * HW * 144 * 4);
  __bf16* pw0 = (__bf16*)carve((size_t)4  * 63 * 512 * 2);  // K=2016, N=64
  __bf16* pw1 = (__bf16*)carve((size_t)4  * 18 * 512 * 2);  // K=576,  N=64
  __bf16* pw2 = (__bf16*)carve((size_t)4  * 18 * 512 * 2);
  __bf16* pw3 = (__bf16*)carve((size_t)28 * 18 * 512 * 2);  // K=576,  N=432(+pad)
  __bf16* pwd = (__bf16*)carve((size_t)4  * 36 * 512 * 2);  // K=1152, N=64

  // Stage feature tensor + repack all weights into WMMA fragment order.
  k_pack_feat<<<dim3(4096), 256, 0, stream>>>(extra, f1, f2, feat);
  k_repack<<<dim3(512), 256, 0, stream>>>(ow0, pw0, 64,  COFF, COFFP, 4,  63, 0);
  k_repack<<<dim3(512), 256, 0, stream>>>(ow1, pw1, 64,  64,   64,    4,  18, 0);
  k_repack<<<dim3(512), 256, 0, stream>>>(ow2, pw2, 64,  64,   64,    4,  18, 0);
  k_repack<<<dim3(512), 256, 0, stream>>>(ow3, pw3, C3O, 64,   64,    28, 18, 0);
  k_repack<<<dim3(512), 256, 0, stream>>>(wgt, pwd, 64,  CIN,  0,     4,  36, 1);

  // Offset-network convs (WMMA implicit GEMM).
  k_conv_lrelu<COFFP, 63><<<dim3(B_ * H_ * 3), 128, 3 * 36 * COFFP * 2, stream>>>(
      feat, pw0, ob0, h1);
  k_conv_lrelu<64, 18><<<dim3(B_ * H_ * 3), 128, 3 * 36 * 64 * 2, stream>>>(
      h1, pw1, ob1, h2);
  k_conv_lrelu<64, 18><<<dim3(B_ * H_ * 3), 128, 3 * 36 * 64 * 2, stream>>>(
      h2, pw2, ob2, h3);
  k_conv_offsets<<<dim3(B_ * H_ * 3, 7), 128, 3 * 36 * 64 * 2, stream>>>(
      h3, pw3, ob3, f1, f2, offsb, maskb);

  // Deformable conv (sample -> LDS -> WMMA), M=32 tile, 72 KB LDS.
  k_deform<<<dim3(B_ * H_ * 3), 128, 32 * KD * 2, stream>>>(
      x, offsb, maskb, pwd, bias, out);
  (void)in_sizes; (void)n_in; (void)out_size; (void)ws_size;
}
